// VRNN_62689342652472
// MI455X (gfx1250) — compile-verified, matrix operand order checked by reading, not run on hardware
//
#include <hip/hip_runtime.h>

typedef __attribute__((ext_vector_type(16))) _Float16 v16h;
typedef __attribute__((ext_vector_type(8)))  float    v8f;

#define KH    16     // hidden size k
#define ST    8      // state size
#define LDCAT 68     // cat: cols 0-15 xe | 16-31 ue | 32-47 z_e | 48-63 h
#define LDT   52     // gated-t buffer [16][48]
#define LDS16 20     // 16x16 staging stride (bank-conflict-free, 16B-aligned rows)
#define NWAVE 2      // waves per block
#define NOPS  22     // pre-packed f16 WMMA B-operands

// operand ids in the LDS weight arena (each operand = 32 lanes x 32B = 1KB)
enum { OP_U1 = 0, OP_U2, OP_X1, OP_X2, OP_Z1, OP_Z2,
       OP_G1 = 6,            // 6..11  : post_W1, idx = OP_G1 + 2*j + c
       OP_G2 = 12,           // 12..17 : post_W2
       OP_H0 = 18, OP_H1 = 19, OP_RLO = 20, OP_RHI = 21 };

static __device__ __forceinline__ v8f wmma_f16(v16h a, v16h b, v8f c) {
  return __builtin_amdgcn_wmma_f32_16x16x32_f16(false, a, false, b, (short)0, c,
                                                false, false);
}

// native gfx1250 v_tanh_f32 when available
static __device__ __forceinline__ float ftanh(float x) {
#if __has_builtin(__builtin_amdgcn_tanhf)
  return __builtin_amdgcn_tanhf(x);
#else
  return tanhf(x);
#endif
}
// sigmoid(x) = 0.5*tanh(0.5x) + 0.5  (1 trans + 1 fma, no rcp)
static __device__ __forceinline__ float fsigmoid(float x) {
  return fmaf(ftanh(0.5f * x), 0.5f, 0.5f);
}

// A operand (16x32 f16) from LDS row-major f32. Halves 8g..8g+7 hold
// K = kbase + 16*g + hi*8 + i (ISA 16-bit A layout). kbase/kcols literals at
// call sites, chosen so the bound test is lane-uniform (no divergence).
static __device__ __forceinline__ v16h make_a_lds(const float* buf, int ld,
                                                  int kbase, int kcols, int lane) {
  const int m = lane & 15, hi = lane >> 4;
  const float* row = buf + m * ld;
  v16h a;
#pragma unroll
  for (int g = 0; g < 2; ++g) {
    const int k0 = kbase + 16 * g + hi * 8;
#pragma unroll
    for (int i = 0; i < 8; ++i) {
      const int k = k0 + i;
      a[8 * g + i] = (k < kcols) ? (_Float16)row[k] : (_Float16)0.0f;
    }
  }
  return a;
}

// ---- one-time B-operand builders (global f32 -> packed f16 B layout) ----
static __device__ __forceinline__ v16h make_b_glb(const float* W, int ld, int kbase,
                                                  int krows, int nbase, int lane) {
  const int n = nbase + (lane & 15), hi = lane >> 4;
  v16h b;
#pragma unroll
  for (int g = 0; g < 2; ++g) {
#pragma unroll
    for (int i = 0; i < 8; ++i) {
      const int k = kbase + 16 * g + hi * 8 + i;
      b[8 * g + i] = (k < krows) ? (_Float16)W[k * ld + n] : (_Float16)0.0f;
    }
  }
  return b;
}

static __device__ __forceinline__ v16h make_b_head(const float* Wmu, const float* Wls,
                                                   int kbase, int lane) {
  const int n = lane & 15, hi = lane >> 4;
  const float* W = (n < 8) ? Wmu : Wls;
  const int nn = n & 7;
  v16h b;
#pragma unroll
  for (int g = 0; g < 2; ++g) {
#pragma unroll
    for (int i = 0; i < 8; ++i) {
      const int k = kbase + 16 * g + hi * 8 + i;
      b[8 * g + i] = (k < 48) ? (_Float16)W[k * 8 + nn] : (_Float16)0.0f;
    }
  }
  return b;
}

static __device__ __forceinline__ v16h make_b_rnnhi(const float* Wih, const float* Whh,
                                                    int lane) {
  const int n = lane & 15, hi = lane >> 4;
  v16h b;
#pragma unroll
  for (int g = 0; g < 2; ++g) {
#pragma unroll
    for (int i = 0; i < 8; ++i) {
      const int kl = 16 * g + hi * 8 + i;
      const float v = (kl < 16) ? Wih[(32 + kl) * 16 + n] : Whh[(kl - 16) * 16 + n];
      b[8 * g + i] = (_Float16)v;
    }
  }
  return b;
}

// scatter C/D tile (lane=N, VGPR r -> M = r + 8*hi) into LDS row-major
static __device__ __forceinline__ void store_c_lds(float* buf, int ld, int col0,
                                                   v8f c, int lane) {
  const int n = lane & 15, hi = lane >> 4;
#pragma unroll
  for (int r = 0; r < 8; ++r) buf[(r + 8 * hi) * ld + col0 + n] = c[r];
}

// two independent two-layer Linear+ReLU chains, interleaved so the second
// WMMA/valu of one chain fills the other's co-exec hazard window.
static __device__ __forceinline__ void pre2x2(v16h aU, v16h B1U, float b1U, v16h B2U, float b2U, float* tmpU,
                                              v16h aX, v16h B1X, float b1X, v16h B2X, float b2X, float* tmpX,
                                              v8f& outU, v8f& outX, int lane) {
  v8f cU = {}; cU = wmma_f16(aU, B1U, cU);
  v8f cX = {}; cX = wmma_f16(aX, B1X, cX);
#pragma unroll
  for (int r = 0; r < 8; ++r) {
    cU[r] = fmaxf(cU[r] + b1U, 0.0f);
    cX[r] = fmaxf(cX[r] + b1X, 0.0f);
  }
  store_c_lds(tmpU, LDS16, 0, cU, lane);
  store_c_lds(tmpX, LDS16, 0, cX, lane);
  v16h a2U = make_a_lds(tmpU, LDS16, 0, 16, lane);
  v16h a2X = make_a_lds(tmpX, LDS16, 0, 16, lane);
  v8f dU = {}; dU = wmma_f16(a2U, B2U, dU);
  v8f dX = {}; dX = wmma_f16(a2X, B2X, dX);
#pragma unroll
  for (int r = 0; r < 8; ++r) {
    dU[r] = fmaxf(dU[r] + b2U, 0.0f);
    dX[r] = fmaxf(dX[r] + b2X, 0.0f);
  }
  outU = dU; outX = dX;
}

// single two-layer Linear+ReLU (z path)
static __device__ __forceinline__ v8f pre2(v16h a1, v16h B1, float b1n, v16h B2,
                                           float b2n, float* tmp, int lane) {
  v8f c = {};
  c = wmma_f16(a1, B1, c);
#pragma unroll
  for (int r = 0; r < 8; ++r) c[r] = fmaxf(c[r] + b1n, 0.0f);
  store_c_lds(tmp, LDS16, 0, c, lane);
  v16h a2 = make_a_lds(tmp, LDS16, 0, 16, lane);
  v8f c2 = {};
  c2 = wmma_f16(a2, B2, c2);
#pragma unroll
  for (int r = 0; r < 8; ++r) c2[r] = fmaxf(c2[r] + b2n, 0.0f);
  return c2;
}

__global__ __launch_bounds__(32 * NWAVE) void vrnn_scan(
    const float* __restrict__ ext, const float* __restrict__ obs,
    const float* __restrict__ eps,
    const float* pu_W1, const float* pu_b1, const float* pu_W2, const float* pu_b2,
    const float* px_W1, const float* px_b1, const float* px_W2, const float* px_b2,
    const float* pz_W1, const float* pz_b1, const float* pz_W2, const float* pz_b2,
    const float* post_W1, const float* post_b1, const float* post_W2, const float* post_b2,
    const float* post_Wmu, const float* post_bmu, const float* post_Wls, const float* post_bls,
    const float* rnn_Wih, const float* rnn_bih, const float* rnn_Whh, const float* rnn_bhh,
    float* __restrict__ out, int L, int B) {
  const int lane = threadIdx.x & 31;
  const int wv   = threadIdx.x >> 5;
  const int gw   = blockIdx.x * NWAVE + wv;
  const int r0   = gw * 16;                       // this wave's 16 batch rows

  // ---- LDS arenas ----
  __shared__ __align__(16) float wlds[NOPS * 256];      // packed f16 B-operands
  constexpr int PER_WAVE = 16 * LDCAT + 16 * LDT + 6 * 16 * LDS16;
  __shared__ __align__(16) float dsm[NWAVE * PER_WAVE]; // per-wave data buffers
  float* base = dsm + wv * PER_WAVE;
  float* cat  = base;                 // [16][68]
  float* tbuf = cat  + 16 * LDCAT;    // [16][52]
  float* tmpA = tbuf + 16 * LDT;      // [16][20] stage-1 scratch (u / z path)
  float* tmpB = tmpA + 16 * LDS16;    // [16][20] stage-1 scratch (x path)
  float* muls = tmpB + 16 * LDS16;    // [16][20] mu | logsigma
  float* zbuf = muls + 16 * LDS16;    // [16][20] z (cols 8..15 zero)
  float* stgu = zbuf + 16 * LDS16;    // [16][20] raw ext (cols 8..15 zero)
  float* stgx = stgu + 16 * LDS16;    // [16][20] raw obs (cols 4..15 zero)

  const int n  = lane & 15;
  const int hi = lane >> 4;

  // ---- one-time: pack all WMMA B-operands into LDS (split across waves) ----
  for (int op = wv; op < NOPS; op += NWAVE) {
    v16h b;
    if      (op == OP_U1) b = make_b_glb(pu_W1, 16, 0, 8,  0, lane);
    else if (op == OP_U2) b = make_b_glb(pu_W2, 16, 0, 16, 0, lane);
    else if (op == OP_X1) b = make_b_glb(px_W1, 16, 0, 4,  0, lane);
    else if (op == OP_X2) b = make_b_glb(px_W2, 16, 0, 16, 0, lane);
    else if (op == OP_Z1) b = make_b_glb(pz_W1, 16, 0, 8,  0, lane);
    else if (op == OP_Z2) b = make_b_glb(pz_W2, 16, 0, 16, 0, lane);
    else if (op < OP_G2)  { const int q = op - OP_G1;
                            b = make_b_glb(post_W1, 48, 32 * (q & 1), 48, 16 * (q >> 1), lane); }
    else if (op < OP_H0)  { const int q = op - OP_G2;
                            b = make_b_glb(post_W2, 48, 32 * (q & 1), 48, 16 * (q >> 1), lane); }
    else if (op == OP_H0) b = make_b_head(post_Wmu, post_Wls, 0,  lane);
    else if (op == OP_H1) b = make_b_head(post_Wmu, post_Wls, 32, lane);
    else if (op == OP_RLO) b = make_b_glb(rnn_Wih, 16, 0, 48, 0, lane);
    else                   b = make_b_rnnhi(rnn_Wih, rnn_Whh, lane);
    *(v16h*)(wlds + op * 256 + lane * 8) = b;
  }

  // zero pads + h0 (persist across the whole loop)
  for (int i = lane; i < 256; i += 32) cat[(i >> 4) * LDCAT + 48 + (i & 15)] = 0.0f;
  {
    const int m = lane >> 1, c4 = 8 + (lane & 1) * 4;
    *(float4*)(stgu + m * LDS16 + c4) = float4{0.f, 0.f, 0.f, 0.f};
    *(float4*)(zbuf + m * LDS16 + c4) = float4{0.f, 0.f, 0.f, 0.f};
  }
  for (int i = lane; i < 16 * 12; i += 32) stgx[(i / 12) * LDS16 + 4 + (i % 12)] = 0.0f;
  __syncthreads();

#define LDOP(idx) (*(const v16h*)(wlds + (idx) * 256 + lane * 8))

  // per-lane (column) biases
  const float ub1 = pu_b1[n], ub2 = pu_b2[n];
  const float xb1 = px_b1[n], xb2 = px_b2[n];
  const float zb1 = pz_b1[n], zb2 = pz_b2[n];
  float gb1[3], gb2[3];
#pragma unroll
  for (int j = 0; j < 3; ++j) { gb1[j] = post_b1[16 * j + n]; gb2[j] = post_b2[16 * j + n]; }
  const float hbn = (n < 8) ? post_bmu[n] : post_bls[n - 8];
  const float rbn = rnn_bih[n] + rnn_bhh[n];

  const int LB8 = L * B * ST;
  float* out_mu = out;
  float* out_ls = out + LB8;
  float* out_h  = out + 2 * LB8;

  for (int t = 0; t < L; ++t) {
    const float* ext_t = ext + (t * B + r0) * 8;
    const float* obs_t = obs + (t * B + r0) * 4;
    const float* eps_t = eps + (t * B + r0) * 8;
    if (t + 1 < L) {   // hide next step's only critical-path global loads
      __builtin_prefetch(ext + ((t + 1) * B + r0) * 8, 0, 1);
      __builtin_prefetch(obs + ((t + 1) * B + r0) * 4, 0, 1);
      __builtin_prefetch(eps + ((t + 1) * B + r0) * 8, 0, 1);
    }

    // stage raw inputs (tiles are contiguous: 128 / 64 floats), coalesced
    {
      const float4 v = *(const float4*)(ext_t + lane * 4);
      *(float4*)(stgu + (lane >> 1) * LDS16 + (lane & 1) * 4) = v;
      const float2 w = *(const float2*)(obs_t + lane * 2);
      *(float2*)(stgx + (lane >> 1) * LDS16 + (lane & 1) * 2) = w;
    }

    // u_e -> cat[:,16..31] ; x_e -> cat[:,0..15]   (interleaved chains)
    {
      v16h aU = make_a_lds(stgu, LDS16, 0, 16, lane);
      v16h aX = make_a_lds(stgx, LDS16, 0, 16, lane);
      v8f ue, xe;
      pre2x2(aU, LDOP(OP_U1), ub1, LDOP(OP_U2), ub2, tmpA,
             aX, LDOP(OP_X1), xb1, LDOP(OP_X2), xb2, tmpB, ue, xe, lane);
      store_c_lds(cat, LDCAT, 16, ue, lane);
      store_c_lds(cat, LDCAT, 0,  xe, lane);
    }

    // gated hidden: t = tanh(cat@W1+b1) * sigmoid(cat@W2+b2), cat = [xe|ue|h]
    v16h a_lo = make_a_lds(cat, LDCAT, 0,  64, lane);   // K 0..31  (xe|ue)
    v16h a_h  = make_a_lds(cat, LDCAT, 48, 64, lane);   // K 32..47 (h), pad 0
#pragma unroll
    for (int j = 0; j < 3; ++j) {
      v8f c1 = {};
      c1 = wmma_f16(a_lo, LDOP(OP_G1 + 2 * j),     c1);
      c1 = wmma_f16(a_h,  LDOP(OP_G1 + 2 * j + 1), c1);
      v8f c2 = {};
      c2 = wmma_f16(a_lo, LDOP(OP_G2 + 2 * j),     c2);
      c2 = wmma_f16(a_h,  LDOP(OP_G2 + 2 * j + 1), c2);
      v8f tv;
#pragma unroll
      for (int r = 0; r < 8; ++r)
        tv[r] = ftanh(c1[r] + gb1[j]) * fsigmoid(c2[r] + gb2[j]);
      store_c_lds(tbuf, LDT, 16 * j, tv, lane);
    }

    // heads: [mu | logsigma] in one 16x16 tile
    v16h t_lo = make_a_lds(tbuf, LDT, 0,  48, lane);
    v16h t_hi = make_a_lds(tbuf, LDT, 32, 48, lane);
    v8f hc = {};
    hc = wmma_f16(t_lo, LDOP(OP_H0), hc);
    hc = wmma_f16(t_hi, LDOP(OP_H1), hc);
#pragma unroll
    for (int r = 0; r < 8; ++r) hc[r] += hbn;
    {
      float* dst = (n < 8) ? out_mu : out_ls;
      const int col   = n & 7;
      const int obase = (t * B + r0) * ST;
#pragma unroll
      for (int r = 0; r < 8; ++r) dst[obase + (r + 8 * hi) * ST + col] = hc[r];
    }
    store_c_lds(muls, LDS16, 0, hc, lane);

    // z = mu + exp(0.5*ls)*eps  (lane -> row lane>>1, 4 state cols)
    {
      const int m  = lane >> 1;
      const int s0 = (lane & 1) * 4;
      const float4 e = *(const float4*)(eps_t + m * 8 + s0);
      const float* mr = muls + m * LDS16;
      float* zr = zbuf + m * LDS16 + s0;
      zr[0] = fmaf(__expf(0.5f * mr[8 + s0 + 0]), e.x, mr[s0 + 0]);
      zr[1] = fmaf(__expf(0.5f * mr[8 + s0 + 1]), e.y, mr[s0 + 1]);
      zr[2] = fmaf(__expf(0.5f * mr[8 + s0 + 2]), e.z, mr[s0 + 2]);
      zr[3] = fmaf(__expf(0.5f * mr[8 + s0 + 3]), e.w, mr[s0 + 3]);
    }

    // z_e -> cat[:,32..47]
    {
      v16h a = make_a_lds(zbuf, LDS16, 0, 16, lane);
      store_c_lds(cat, LDCAT, 32, pre2(a, LDOP(OP_Z1), zb1, LDOP(OP_Z2), zb2, tmpA, lane), lane);
    }

    // RNN: h_new = tanh([xe|ue|z_e]@Wih + h@Whh + b)   (Whh folded into chunk 2)
    v16h a_hi2 = make_a_lds(cat, LDCAT, 32, 64, lane);  // K 32..63 = [z_e | h]
    v8f rc = {};
    rc = wmma_f16(a_lo,  LDOP(OP_RLO), rc);
    rc = wmma_f16(a_hi2, LDOP(OP_RHI), rc);
#pragma unroll
    for (int r = 0; r < 8; ++r) rc[r] = ftanh(rc[r] + rbn);
    store_c_lds(cat, LDCAT, 48, rc, lane);              // h <- h_new
  }

  // final hidden state [B,16]
#pragma unroll
  for (int r = 0; r < 8; ++r) {
    const int m = r + 8 * hi;
    out_h[(r0 + m) * KH + n] = cat[m * LDCAT + 48 + n];
  }
#undef LDOP
}

extern "C" void kernel_launch(void* const* d_in, const int* in_sizes, int n_in,
                              void* d_out, int out_size, void* d_ws, size_t ws_size,
                              hipStream_t stream) {
  const float* ext = (const float*)d_in[0];
  const float* obs = (const float*)d_in[1];
  const float* eps = (const float*)d_in[2];
  const float* pu_W1 = (const float*)d_in[3];
  const float* pu_b1 = (const float*)d_in[4];
  const float* pu_W2 = (const float*)d_in[5];
  const float* pu_b2 = (const float*)d_in[6];
  const float* px_W1 = (const float*)d_in[7];
  const float* px_b1 = (const float*)d_in[8];
  const float* px_W2 = (const float*)d_in[9];
  const float* px_b2 = (const float*)d_in[10];
  const float* pz_W1 = (const float*)d_in[11];
  const float* pz_b1 = (const float*)d_in[12];
  const float* pz_W2 = (const float*)d_in[13];
  const float* pz_b2 = (const float*)d_in[14];
  const float* post_W1 = (const float*)d_in[15];
  const float* post_b1 = (const float*)d_in[16];
  const float* post_W2 = (const float*)d_in[17];
  const float* post_b2 = (const float*)d_in[18];
  const float* post_Wmu = (const float*)d_in[19];
  const float* post_bmu = (const float*)d_in[20];
  const float* post_Wls = (const float*)d_in[21];
  const float* post_bls = (const float*)d_in[22];
  const float* rnn_Wih = (const float*)d_in[23];
  const float* rnn_bih = (const float*)d_in[24];
  const float* rnn_Whh = (const float*)d_in[25];
  const float* rnn_bhh = (const float*)d_in[26];

  const int B = 1024;
  const int L = in_sizes[0] / (B * 8);   // external_input_seq = [L,B,8]

  const int waves  = B / 16;             // 64 independent recurrence tiles
  const int blocks = waves / NWAVE;      // 2 waves (64 threads) per block
  vrnn_scan<<<blocks, 32 * NWAVE, 0, stream>>>(
      ext, obs, eps,
      pu_W1, pu_b1, pu_W2, pu_b2,
      px_W1, px_b1, px_W2, px_b2,
      pz_W1, pz_b1, pz_W2, pz_b2,
      post_W1, post_b1, post_W2, post_b2,
      post_Wmu, post_bmu, post_Wls, post_bls,
      rnn_Wih, rnn_bih, rnn_Whh, rnn_bhh,
      (float*)d_out, L, B);
}
